// BiLSTM_52304111730944
// MI455X (gfx1250) — matrix operand order repaired and checked by reading, not compiled, not run on hardware
//
#include <hip/hip_runtime.h>
#include <hip/hip_bf16.h>
#include <cstdint>
#include <cstddef>

typedef __attribute__((ext_vector_type(16))) __bf16 v16bf;
typedef __attribute__((ext_vector_type(8)))  float  v8f;

#define Hn 64
#define Gn 256   /* 4*H */
#define Tn 512
#define Bn 1024

__device__ __forceinline__ unsigned short f2bf(float f) {
    union { float f; uint32_t u; } v; v.f = f;
    uint32_t u = v.u;
    return (unsigned short)((u + 0x7FFFu + ((u >> 16) & 1u)) >> 16);  // RNE
}

// Pack two floats into one DWORD of bf16 pair (lo in [15:0], hi in [31:16]).
__device__ __forceinline__ uint32_t pack2bf(float lo, float hi) {
    union { float f; uint32_t u; } a, b; a.f = lo; b.f = hi;
    const uint32_t ua = a.u + 0x7FFFu + ((a.u >> 16) & 1u);
    const uint32_t ub = b.u + 0x7FFFu + ((b.u >> 16) & 1u);
    return (ua >> 16) | (ub & 0xFFFF0000u);
}

union Frag16 { v16bf v; uint4 q[2]; uint32_t w[8]; unsigned short s[16]; };

__device__ __forceinline__ v8f wmma_bf16(v16bf a, v16bf b, v8f c) {
    return __builtin_amdgcn_wmma_f32_16x16x32_bf16(false, a, false, b, (short)0, c, false, false);
}

__device__ __forceinline__ float rcp_fast(float x) {
#if __has_builtin(__builtin_amdgcn_rcpf)
    return __builtin_amdgcn_rcpf(x);      // v_rcp_f32
#else
    return 1.0f / x;
#endif
}
__device__ __forceinline__ float sigm(float x) {
    return rcp_fast(1.0f + __expf(-x));                 // exp, add, rcp
}
__device__ __forceinline__ float tanh_f(float x) {
    // tanh(x) = 2*sigmoid(2x) - 1; saturates correctly (exp->inf => rcp->0 => -1)
    return __builtin_fmaf(2.0f, rcp_fast(1.0f + __expf(-2.0f * x)), -1.0f);
}

// B fragment (K=32 x N=16) from row-major f32 weight W[g][k].
// Lane L holds column n = L&15, K = kb*32 + (L>>4)*8 + 0..7 and kb*32 + 16 + (L>>4)*8 + 0..7.
template<int LDK>
__device__ __forceinline__ v16bf loadBfrag(const float* __restrict__ W, int n0, int kb) {
    const int lane = threadIdx.x & 31;
    const float* p = W + (size_t)(n0 + (lane & 15)) * LDK + kb * 32;
    const int hb = (lane >> 4) << 3;
    Frag16 f;
#pragma unroll
    for (int i = 0; i < 8; ++i) { const int c = (i < 4) ? (hb + 2 * i) : (16 + hb + 2 * (i - 4)); f.w[i] = pack2bf(p[c], p[c + 1]); }
    return f.v;
}

// A fragment (M=16 x K=32 block kb) from LDS bf16 tile, row-major [16][LDK].
template<int LDK>
__device__ __forceinline__ v16bf loadAfrag(const unsigned short* tile, int kb) {
    const int lane = threadIdx.x & 31;
    const int m = lane & 15;
    const int base = kb * 32 + ((lane >> 4) << 3);
    Frag16 f;
    f.q[0] = *(const uint4*)(tile + m * LDK + base);
    f.q[1] = *(const uint4*)(tile + m * LDK + base + 16);
    return f.v;
}

// x A-fragment (16x32 f32 tile) straight from global into registers.
// Lane L = row m = L&15; K = hi*8 + 0..7 and 16 + hi*8 + 0..7.
struct XRaw { float4 a, b, c, d; };
__device__ __forceinline__ XRaw loadXraw(const float* __restrict__ x, int bg0, int ts) {
    const int lane = threadIdx.x & 31;
    const int m = lane & 15;
    const int base = (lane >> 4) << 3;
    const float* p = x + (size_t)(bg0 + m) * Tn * 32 + (size_t)ts * 32 + base;
    XRaw r;
    r.a = *(const float4*)(p);
    r.b = *(const float4*)(p + 4);
    r.c = *(const float4*)(p + 16);
    r.d = *(const float4*)(p + 20);
    return r;
}
__device__ __forceinline__ v16bf cvtXfrag(const XRaw& r) {
    Frag16 f;
    const float* s = &r.a.x;
#pragma unroll
    for (int i = 0; i < 8; ++i) f.w[i] = pack2bf(s[2 * i], s[2 * i + 1]);
    return f.v;
}

// h1 A-fragment (16x128 bf16 tile) straight from global into registers.
__device__ __forceinline__ v16bf loadInFrag(const unsigned short* __restrict__ h1,
                                            int bg0, int ts, int kb) {
    const int lane = threadIdx.x & 31;
    const int m = lane & 15;
    const int base = kb * 32 + ((lane >> 4) << 3);
    const unsigned short* p = h1 + (size_t)ts * Bn * 128 + (size_t)(bg0 + m) * 128 + base;
    Frag16 f;
    f.q[0] = *(const uint4*)(p);
    f.q[1] = *(const uint4*)(p + 16);
    return f.v;
}

// ---------------- Layer 0: pipelined bidirectional scan ------------------------
// grid = 128 (dir, batch-chunk), block = 128 (4 waves; wave jg owns i/f/g/o
// tiles of j-columns [jg*16, jg*16+16)). Input projection for step t+1 is
// computed during step t; loop-carried path = ds_load Ah -> 2 WMMA -> cell ->
// h store -> 1 barrier.
__global__ __launch_bounds__(128) void lstm_scan_l0(
    const float* __restrict__ x,      // (B, T, 32)
    const float* __restrict__ w_ih,   // (2, 256, 32)
    const float* __restrict__ w_hh,   // (2, 256, 64)
    const float* __restrict__ b_ih,   // (2, 256)
    const float* __restrict__ b_hh,   // (2, 256)
    unsigned short* __restrict__ h1)  // (T, B, 128) bf16
{
    __shared__ alignas(16) unsigned short hbuf[2][16 * 64];

    const int bid  = blockIdx.x;
    const int dir  = bid >> 6;
    const int bc   = bid & 63;
    const int tid  = threadIdx.x;
    const int jg   = tid >> 5;
    const int lane = tid & 31;
    const int n    = lane & 15;
    const int hi   = lane >> 4;
    const int bg0  = bc * 16;

    const float* Wih = w_ih + (size_t)dir * Gn * 32;
    const float* Whh = w_hh + (size_t)dir * Gn * Hn;

    v16bf Bi[4], Bh[4][2];
    v8f cb[4];
#pragma unroll
    for (int q = 0; q < 4; ++q) {
        const int n0 = q * 64 + jg * 16;
        Bi[q]    = loadBfrag<32>(Wih, n0, 0);
        Bh[q][0] = loadBfrag<64>(Whh, n0, 0);
        Bh[q][1] = loadBfrag<64>(Whh, n0, 1);
        const float bv = b_ih[dir * Gn + n0 + n] + b_hh[dir * Gn + n0 + n];
#pragma unroll
        for (int r = 0; r < 8; ++r) cb[q][r] = bv;
    }

    v8f cst;
#pragma unroll
    for (int r = 0; r < 8; ++r) cst[r] = 0.f;
    for (int i = tid; i < 16 * 64; i += 128) hbuf[0][i] = 0;   // h0 = 0
    int cur = 0;

    // Prologue: input projection for t = 0
    v8f pacc[4];
    {
        const int ts0 = dir ? (Tn - 1) : 0;
        const v16bf Ax0 = cvtXfrag(loadXraw(x, bg0, ts0));
#pragma unroll
        for (int q = 0; q < 4; ++q) pacc[q] = wmma_bf16(Ax0, Bi[q], cb[q]);
    }
    __syncthreads();

    for (int t = 0; t < Tn; ++t) {
        const int ts = dir ? (Tn - 1 - t) : t;
        const int tn = (t + 1 < Tn) ? (t + 1) : 0;          // clamped (uniform)
        const int tsn = dir ? (Tn - 1 - tn) : tn;

        // Issue next-step x loads immediately (latency hidden under compute)
        const XRaw xraw = loadXraw(x, bg0, tsn);

        // Loop-carried recurrence
        const v16bf Ah0 = loadAfrag<64>(hbuf[cur], 0);
        const v16bf Ah1 = loadAfrag<64>(hbuf[cur], 1);
        v8f acc[4];
#pragma unroll
        for (int q = 0; q < 4; ++q)
            acc[q] = wmma_bf16(Ah1, Bh[q][1], wmma_bf16(Ah0, Bh[q][0], pacc[q]));

        // Cell update (register resident); D-layout: m = r + hi*8, col jg*16+n
        unsigned short* hb = hbuf[cur ^ 1];
#pragma unroll
        for (int r = 0; r < 8; ++r) {
            const float ig = sigm(acc[0][r]);
            const float fg = sigm(acc[1][r]);
            const float gg = tanh_f(acc[2][r]);
            const float og = sigm(acc[3][r]);
            const float c  = fg * cst[r] + ig * gg;
            cst[r] = c;
            const float h = og * tanh_f(c);
            const unsigned short us = f2bf(h);
            const int m = r + (hi << 3);
            const int j = jg * 16 + n;
            hb[m * 64 + j] = us;
            h1[(size_t)ts * Bn * 128 + (size_t)(bg0 + m) * 128 + dir * 64 + j] = us;
        }

        // Off-path: input projection for t+1
        const v16bf Axn = cvtXfrag(xraw);
#pragma unroll
        for (int q = 0; q < 4; ++q) pacc[q] = wmma_bf16(Axn, Bi[q], cb[q]);

        __syncthreads();
        cur ^= 1;
    }
}

// ---------------- Layer 1: pipelined scan over h1 (K=128). Backward = 1 step. --
__global__ __launch_bounds__(128) void lstm_scan_l1(
    const unsigned short* __restrict__ h1, // (T, B, 128) bf16
    const float* __restrict__ w_ih,        // (2, 256, 128)
    const float* __restrict__ w_hh,        // (2, 256, 64)
    const float* __restrict__ b_ih,
    const float* __restrict__ b_hh,
    float* __restrict__ h2last)            // (B, 128) f32
{
    __shared__ alignas(16) unsigned short hbuf[2][16 * 64];

    const int bid  = blockIdx.x;
    const int dir  = bid >> 6;
    const int bc   = bid & 63;
    const int tid  = threadIdx.x;
    const int jg   = tid >> 5;
    const int lane = tid & 31;
    const int n    = lane & 15;
    const int hi   = lane >> 4;
    const int bg0  = bc * 16;

    const float* Wih = w_ih + (size_t)dir * Gn * 128;
    const float* Whh = w_hh + (size_t)dir * Gn * Hn;

    v16bf Bi[4][4], Bh[4][2];
    v8f cb[4];
#pragma unroll
    for (int q = 0; q < 4; ++q) {
        const int n0 = q * 64 + jg * 16;
#pragma unroll
        for (int kb = 0; kb < 4; ++kb) Bi[q][kb] = loadBfrag<128>(Wih, n0, kb);
        Bh[q][0] = loadBfrag<64>(Whh, n0, 0);
        Bh[q][1] = loadBfrag<64>(Whh, n0, 1);
        const float bv = b_ih[dir * Gn + n0 + n] + b_hh[dir * Gn + n0 + n];
#pragma unroll
        for (int r = 0; r < 8; ++r) cb[q][r] = bv;
    }

    v8f cst;
#pragma unroll
    for (int r = 0; r < 8; ++r) cst[r] = 0.f;
    for (int i = tid; i < 16 * 64; i += 128) hbuf[0][i] = 0;
    int cur = 0;

    // Output only needs fwd step T-1 and bwd step 0 -> bwd runs one step.
    const int nsteps = dir ? 1 : Tn;

    // Prologue: input projection for t = 0
    v8f pacc[4];
    {
        const int ts0 = dir ? (Tn - 1) : 0;
        v16bf A0[4];
#pragma unroll
        for (int kb = 0; kb < 4; ++kb) A0[kb] = loadInFrag(h1, bg0, ts0, kb);
#pragma unroll
        for (int q = 0; q < 4; ++q) {
            v8f a = cb[q];
#pragma unroll
            for (int kb = 0; kb < 4; ++kb) a = wmma_bf16(A0[kb], Bi[q][kb], a);
            pacc[q] = a;
        }
    }
    __syncthreads();

    for (int t = 0; t < nsteps; ++t) {
        const int tn = (t + 1 < nsteps) ? (t + 1) : 0;      // clamped (uniform)
        const int tsn = dir ? (Tn - 1 - tn) : tn;

        // Issue next-step input loads immediately
        v16bf An[4];
#pragma unroll
        for (int kb = 0; kb < 4; ++kb) An[kb] = loadInFrag(h1, bg0, tsn, kb);

        // Loop-carried recurrence
        const v16bf Ah0 = loadAfrag<64>(hbuf[cur], 0);
        const v16bf Ah1 = loadAfrag<64>(hbuf[cur], 1);
        v8f acc[4];
#pragma unroll
        for (int q = 0; q < 4; ++q)
            acc[q] = wmma_bf16(Ah1, Bh[q][1], wmma_bf16(Ah0, Bh[q][0], pacc[q]));

        const bool writeOut = dir ? (t == 0) : (t == Tn - 1);
        unsigned short* hb = hbuf[cur ^ 1];
#pragma unroll
        for (int r = 0; r < 8; ++r) {
            const float ig = sigm(acc[0][r]);
            const float fg = sigm(acc[1][r]);
            const float gg = tanh_f(acc[2][r]);
            const float og = sigm(acc[3][r]);
            const float c  = fg * cst[r] + ig * gg;
            cst[r] = c;
            const float h = og * tanh_f(c);
            const int m = r + (hi << 3);
            const int j = jg * 16 + n;
            hb[m * 64 + j] = f2bf(h);
            if (writeOut)
                h2last[(size_t)(bg0 + m) * 128 + dir * 64 + j] = h;
        }

        // Off-path: input projection for t+1
#pragma unroll
        for (int q = 0; q < 4; ++q) {
            v8f a = cb[q];
#pragma unroll
            for (int kb = 0; kb < 4; ++kb) a = wmma_bf16(An[kb], Bi[q][kb], a);
            pacc[q] = a;
        }

        __syncthreads();
        cur ^= 1;
    }
}

// ---------------- Final FC: (B,128) @ (128,2)^T + b ----------------------------
__global__ __launch_bounds__(256) void fc_kernel(
    const float* __restrict__ h2last, // (B, 128)
    const float* __restrict__ fc_w,   // (2, 128)
    const float* __restrict__ fc_b,   // (2,)
    float* __restrict__ out)          // (B, 2)
{
    const int gid = blockIdx.x * 256 + threadIdx.x;   // 0 .. 2047
    const int b = gid >> 1, o = gid & 1;
    const float* hv = h2last + (size_t)b * 128;
    const float* wv = fc_w + o * 128;
    float s = fc_b[o];
#pragma unroll 8
    for (int k = 0; k < 128; ++k) s += hv[k] * wv[k];
    out[b * 2 + o] = s;
}

extern "C" void kernel_launch(void* const* d_in, const int* in_sizes, int n_in,
                              void* d_out, int out_size, void* d_ws, size_t ws_size,
                              hipStream_t stream) {
    (void)in_sizes; (void)n_in; (void)out_size; (void)ws_size;
    const float* x     = (const float*)d_in[0];
    const float* w_ih0 = (const float*)d_in[1];
    const float* w_hh0 = (const float*)d_in[2];
    const float* b_ih0 = (const float*)d_in[3];
    const float* b_hh0 = (const float*)d_in[4];
    const float* w_ih1 = (const float*)d_in[5];
    const float* w_hh1 = (const float*)d_in[6];
    const float* b_ih1 = (const float*)d_in[7];
    const float* b_hh1 = (const float*)d_in[8];
    const float* fc_w  = (const float*)d_in[9];
    const float* fc_b  = (const float*)d_in[10];

    unsigned short* h1 = (unsigned short*)d_ws;  // (T,B,128) bf16 = 128 MiB (fits L2)
    float* h2last = (float*)((char*)d_ws + (size_t)Tn * Bn * 128 * 2); // (B,128) f32

    lstm_scan_l0<<<dim3(128), dim3(128), 0, stream>>>(x, w_ih0, w_hh0, b_ih0, b_hh0, h1);
    lstm_scan_l1<<<dim3(128), dim3(128), 0, stream>>>(h1, w_ih1, w_hh1, b_ih1, b_hh1, h2last);
    fc_kernel<<<dim3(8), dim3(256), 0, stream>>>(h2last, fc_w, fc_b, (float*)d_out);
}
